// ContrastiveLoss_8624294330599
// MI455X (gfx1250) — matrix operand order repaired and checked by reading, not compiled, not run on hardware
//
#include <hip/hip_runtime.h>
#include <hip/hip_bf16.h>

typedef __attribute__((ext_vector_type(8)))  __bf16 v8bf;
typedef __attribute__((ext_vector_type(16))) __bf16 v16bf;
typedef __attribute__((ext_vector_type(8)))  float  v8f;

#define N_DIM 128              // feature dim (K)
#define TILE  128              // output tile edge per block
#define LDSB  136              // padded LDS row stride in bf16 (272B = 68 banks, conflict-free)
#define EXP_SCALE 2.8853900817779268f   // (1/tau) * log2(e), tau = 0.5

__device__ __forceinline__ __bf16 f2bf(float f) {
  union { float f; unsigned u; } in; in.f = f;
  unsigned u = in.u;
  unsigned r = u + 0x7FFFu + ((u >> 16) & 1u);   // round-to-nearest-even
  union { unsigned short s; __bf16 b; } out;
  out.s = (unsigned short)(r >> 16);
  return out.b;
}

// Raw v_exp_f32: args are bounded (|x| <= ~3), no denormal guard needed.
__device__ __forceinline__ float fast_exp2(float x) {
  return __builtin_amdgcn_exp2f(x);
}

// ---------- Kernel 0: zero the 4 accumulator arrays + output scalar ----------
__global__ void k_zero(float* __restrict__ acc, int n, float* __restrict__ out) {
  int i = blockIdx.x * blockDim.x + threadIdx.x;
  if (i < n) acc[i] = 0.0f;
  if (i == 0) out[0] = 0.0f;
}

// ---------- Kernel 1: L2-normalize rows (D=128), emit bf16 ----------
__global__ __launch_bounds__(256)
void k_normalize(const float* __restrict__ z1, const float* __restrict__ z2,
                 __bf16* __restrict__ o1, __bf16* __restrict__ o2, int N) {
  const float* x = blockIdx.y ? z2 : z1;
  __bf16*      o = blockIdx.y ? o2 : o1;
  int row  = blockIdx.x * (blockDim.x >> 5) + (threadIdx.x >> 5);
  int lane = threadIdx.x & 31;
  if (row >= N) return;
  const float4 v = ((const float4*)(x + (size_t)row * N_DIM))[lane];
  float ss = v.x*v.x + v.y*v.y + v.z*v.z + v.w*v.w;
  ss += __shfl_xor(ss, 1, 32);  ss += __shfl_xor(ss, 2, 32);
  ss += __shfl_xor(ss, 4, 32);  ss += __shfl_xor(ss, 8, 32);
  ss += __shfl_xor(ss, 16, 32);
  float s = 1.0f / fmaxf(sqrtf(ss), 1e-12f);
  union { __bf16 b[4]; uint2 u; } pk;
  pk.b[0] = f2bf(v.x * s); pk.b[1] = f2bf(v.y * s);
  pk.b[2] = f2bf(v.z * s); pk.b[3] = f2bf(v.w * s);
  *(uint2*)(o + (size_t)row * N_DIM + lane * 4) = pk.u;
}

// ---------- Kernel 2: tiled WMMA GEMM + exp + row/col reductions ----------
// grid = (N/128, N/128, 3):  z=0: S11 (sym), z=1: S22 (sym), z=2: S12 (full)
__global__ __launch_bounds__(256)
void k_gemm_exp(const __bf16* __restrict__ z1b, const __bf16* __restrict__ z2b,
                float* __restrict__ rs11, float* __restrict__ rs22,
                float* __restrict__ rs12, float* __restrict__ cs12,
                float* __restrict__ d11,  float* __restrict__ d22,
                float* __restrict__ d12,  int N) {
  const int r = blockIdx.x, c = blockIdx.y, z = blockIdx.z;
  const bool sym = (z < 2);
  if (sym && c < r) return;            // symmetric: upper triangle only (uniform exit)

  const __bf16* A = (z == 1) ? z2b : z1b;
  const __bf16* B = (z == 0) ? z1b : z2b;
  float* rows = (z == 0) ? rs11 : (z == 1) ? rs22 : rs12;
  float* cols = (z == 0) ? rs11 : (z == 1) ? rs22 : cs12;   // transpose credit
  float* diag = (z == 0) ? d11  : (z == 1) ? d22  : d12;
  const bool doTrans = (!sym) || (c != r);   // sym diag tile: no double count
  const bool doDiag  = (r == c);
  const int r0 = r * TILE, c0 = c * TILE;

  __shared__ __bf16 Bs[TILE][LDSB];
  __shared__ float  colAcc[TILE];

  const int tid = threadIdx.x;
  if (tid < TILE) colAcc[tid] = 0.0f;

  // Cooperative stage of B tile (rows c0..c0+127 of B-matrix source) into LDS.
  // 128 rows * 16 chunks (8 bf16 each) = 2048 chunks / 256 threads = 8 per thread.
  for (int i = 0; i < 8; ++i) {
    int chunk = tid + i * 256;
    int row   = chunk >> 4;
    int col8  = (chunk & 15) * 8;
    v8bf v = *(const v8bf*)(B + (size_t)(c0 + row) * N_DIM + col8);
    *(v8bf*)&Bs[row][col8] = v;
  }
  __syncthreads();

  const int wave = tid >> 5, lane = tid & 31;
  const int half = lane >> 4, ln = lane & 15;

  // A fragments: this wave's 16-row strip, all K=128, kept in registers.
  // ISA 16-bit A layout: lanes 0-15 hold K {0..7,16..23}, lanes 16-31 K {8..15,24..31}.
  const int arow = r0 + wave * 16 + ln;
  v16bf afrag[4];
  #pragma unroll
  for (int kb = 0; kb < 4; ++kb) {
    const __bf16* p = A + (size_t)arow * N_DIM + kb * 32 + half * 8;
    v8bf lo = *(const v8bf*)(p);
    v8bf hi = *(const v8bf*)(p + 16);
    afrag[kb] = __builtin_shufflevector(lo, hi, 0,1,2,3,4,5,6,7,8,9,10,11,12,13,14,15);
  }

  float rsum[8];
  #pragma unroll
  for (int j = 0; j < 8; ++j) rsum[j] = 0.0f;

  #pragma unroll
  for (int cg = 0; cg < 8; ++cg) {         // 8 column groups of 16
    v8f acc;
    #pragma unroll
    for (int j = 0; j < 8; ++j) acc[j] = 0.0f;

    #pragma unroll
    for (int kb = 0; kb < 4; ++kb) {
      // ISA 16-bit B layout: lane = column, lanes 0-15 K {0..7,16..23}, lanes 16-31 K {8..15,24..31}.
      const __bf16* p = &Bs[cg * 16 + ln][kb * 32 + half * 8];
      v8bf lo = *(const v8bf*)(p);
      v8bf hi = *(const v8bf*)(p + 16);
      v16bf bfrag = __builtin_shufflevector(lo, hi, 0,1,2,3,4,5,6,7,8,9,10,11,12,13,14,15);
      acc = __builtin_amdgcn_wmma_f32_16x16x32_bf16(
          false, afrag[kb], false, bfrag, (short)0, acc, false, false);
    }

    // C layout: VGPR j holds row (j + 8*half), column ln.
    float e[8];
    float csum = 0.0f;
    #pragma unroll
    for (int j = 0; j < 8; ++j) {
      e[j] = fast_exp2(acc[j] * EXP_SCALE);    // exp(S/tau), raw v_exp_f32
      rsum[j] += e[j];
      csum    += e[j];
    }
    // column partial over this wave's 16 rows: combine the two halves
    csum += __shfl_xor(csum, 16, 32);
    if (doTrans && half == 0) atomicAdd(&colAcc[cg * 16 + ln], csum);

    if (doDiag && cg == wave && ((ln >> 3) == half)) {
      // element with m == n: j = ln & 7, global row = r0 + wave*16 + ln
      diag[r0 + wave * 16 + ln] = e[ln & 7];
    }
  }

  // Row sums over all 128 columns: reduce across the 16 lanes of each half.
  #pragma unroll
  for (int j = 0; j < 8; ++j) {
    float s = rsum[j];
    s += __shfl_xor(s, 1, 32); s += __shfl_xor(s, 2, 32);
    s += __shfl_xor(s, 4, 32); s += __shfl_xor(s, 8, 32);
    if (ln == 0) atomicAdd(&rows[r0 + wave * 16 + j + 8 * half], s);
  }

  __syncthreads();
  if (doTrans && tid < TILE) atomicAdd(&cols[c0 + tid], colAcc[tid]);
}

// ---------- Kernel 3: per-row loss + mean ----------
__global__ __launch_bounds__(256)
void k_finalize(const float* __restrict__ rs11, const float* __restrict__ rs22,
                const float* __restrict__ rs12, const float* __restrict__ cs12,
                const float* __restrict__ d11,  const float* __restrict__ d22,
                const float* __restrict__ d12,  float* __restrict__ out, int N) {
  int i = blockIdx.x * blockDim.x + threadIdx.x;
  float v = 0.0f;
  if (i < N) {
    float den1 = rs11[i] + rs12[i] - d11[i];
    float den2 = rs22[i] + cs12[i] - d22[i];
    float pos  = d12[i];
    const float ln2 = 0.6931471805599453f;
    // 0.5*(-log(pos/den1) - log(pos/den2)) via log2 (raw v_log_f32)
    v = ln2 * (0.5f * (__builtin_amdgcn_logf(den1) + __builtin_amdgcn_logf(den2))
               - __builtin_amdgcn_logf(pos));
  }
  v += __shfl_xor(v, 1, 32);  v += __shfl_xor(v, 2, 32);
  v += __shfl_xor(v, 4, 32);  v += __shfl_xor(v, 8, 32);
  v += __shfl_xor(v, 16, 32);
  __shared__ float sh[8];
  int wave = threadIdx.x >> 5, lane = threadIdx.x & 31;
  if (lane == 0) sh[wave] = v;
  __syncthreads();
  if (threadIdx.x == 0) {
    float s = 0.0f;
    #pragma unroll
    for (int w = 0; w < 8; ++w) s += sh[w];
    atomicAdd(out, s * (1.0f / (float)N));
  }
}

extern "C" void kernel_launch(void* const* d_in, const int* in_sizes, int n_in,
                              void* d_out, int out_size, void* d_ws, size_t ws_size,
                              hipStream_t stream) {
  const float* z1 = (const float*)d_in[0];
  const float* z2 = (const float*)d_in[1];
  const int N = in_sizes[0] / N_DIM;      // 16384

  char* ws = (char*)d_ws;
  const size_t nbElems = (size_t)N * N_DIM;
  __bf16* z1b = (__bf16*)ws;
  __bf16* z2b = z1b + nbElems;
  float* acc  = (float*)(ws + 2 * nbElems * sizeof(__bf16));
  float* rs11 = acc;
  float* rs22 = acc + (size_t)N;
  float* rs12 = acc + (size_t)2 * N;
  float* cs12 = acc + (size_t)3 * N;
  float* d11  = acc + (size_t)4 * N;
  float* d22  = acc + (size_t)5 * N;
  float* d12  = acc + (size_t)6 * N;
  float* out  = (float*)d_out;

  k_zero<<<(4 * N + 255) / 256, 256, 0, stream>>>(acc, 4 * N, out);
  k_normalize<<<dim3((N + 7) / 8, 2), 256, 0, stream>>>(z1, z2, z1b, z2b, N);
  k_gemm_exp<<<dim3(N / TILE, N / TILE, 3), 256, 0, stream>>>(
      z1b, z2b, rs11, rs22, rs12, cs12, d11, d22, d12, N);
  k_finalize<<<(N + 255) / 256, 256, 0, stream>>>(
      rs11, rs22, rs12, cs12, d11, d22, d12, out, N);

  (void)n_in; (void)out_size; (void)ws_size;
}
